// EdgePredictor_SumHead_73830487818332
// MI455X (gfx1250) — matrix operand order repaired and checked by reading, not compiled, not run on hardware
//
#include <hip/hip_runtime.h>

// ---------------------------------------------------------------------------
// EdgePredictor_SumHead for MI455X (gfx1250, wave32, WMMA 16x16x32 f16)
//
//   h1 = relu(x @ W1 + b1)   -> wave-private LDS strip, then register A-frags
//   h2 = relu(h1 @ W2 + b2)  -> never materialized (fused with fc3 dot)
//   s  = h2 @ W3             -> f32 in workspace
//   out[t,i,j] = s[t,i] + s[t,j] + b3   (134 MB NT-store stream)
//
// B matrices (W1^T, W2^T, f16 [N][K]) are staged per 16-column tile into
// double-buffered LDS shared by all 8 waves of the block -> 8x less L2 traffic
// than per-wave global fetch. GEMM2 A-fragments live entirely in VGPRs.
// ---------------------------------------------------------------------------

typedef __attribute__((ext_vector_type(16))) _Float16 v16h;
typedef __attribute__((ext_vector_type(8)))  _Float16 v8h;
typedef __attribute__((ext_vector_type(4)))  _Float16 v4h;
typedef __attribute__((ext_vector_type(8)))  float    v8f;
typedef __attribute__((ext_vector_type(4)))  float    v4f;
typedef __attribute__((ext_vector_type(4)))  unsigned int v4u;

#define LATENT 256
#define SEC    512
#define THIRD  1024
#define TDIM   8
#define NDIM   2048
#define ROWS   (TDIM * NDIM)   // 16384
#define WPB    8               // waves per block
#define RPB    (WPB * 16)      // rows per block = 128

// A-fragment (16x32 f16, ISA 7.12.2): lanes 0-15 hold M=lane, K = kb+[0..7] and
// kb+[16..23]; lanes 16-31 hold M=lane-16, K = kb+[8..15] and kb+[24..31].
// p must already include the per-lane row base and the (lane>>4)*8 K shift.
static __device__ __forceinline__ v16h a_frag(const _Float16* p) {
  v8h lo = *(const v8h*)(p);        // 16B chunk: K offsets +0..7 (or +8..15)
  v8h hi = *(const v8h*)(p + 16);   // 16B chunk: K offsets +16..23 (or +24..31)
  return __builtin_shufflevector(lo, hi, 0,1,2,3,4,5,6,7,8,9,10,11,12,13,14,15);
}

// ---------------------------------------------------------------------------
// Prep: f32 -> f16 conversions (x row-major; W1/W2 transposed to [N][K] so a
// B-fragment is two 8-K runs (kb+8h+[0..7], kb+16+8h+[0..7]) per lane at column N = lane&15).
// ---------------------------------------------------------------------------
__global__ void cvt_x_f16(const float* __restrict__ x, _Float16* __restrict__ xf) {
  int i = blockIdx.x * blockDim.x + threadIdx.x;         // over ROWS*LATENT/4
  v4f v = ((const v4f*)x)[i];
  v4h o = { (_Float16)v[0], (_Float16)v[1], (_Float16)v[2], (_Float16)v[3] };
  ((v4h*)xf)[i] = o;
}

__global__ void cvt_w_transpose(const float* __restrict__ W, _Float16* __restrict__ WT,
                                int K, int N) {
  int idx = blockIdx.x * blockDim.x + threadIdx.x;       // over K*N
  if (idx >= K * N) return;
  int n = idx / K, k = idx - n * K;
  WT[idx] = (_Float16)W[(size_t)k * N + n];              // WT[n*K + k] = W[k][n]
}

// ---------------------------------------------------------------------------
// Fused MLP trunk -> s.
// Block: 256 threads = 8 waves, 128 rows (wave w owns rows 16w..16w+15).
// LDS: h1 strips (8 x 16KB) | B tile buf0 (16KB) | B tile buf1 (16KB).
// ---------------------------------------------------------------------------
__launch_bounds__(WPB * 32, 1)
__global__ void mlp_s_kernel(const _Float16* __restrict__ xf,
                             const _Float16* __restrict__ w1T,
                             const _Float16* __restrict__ w2T,
                             const float* __restrict__ b1,
                             const float* __restrict__ b2,
                             const float* __restrict__ w3,
                             float* __restrict__ s_out) {
  extern __shared__ _Float16 lds[];
  const int tid  = threadIdx.x;
  const int lane = tid & 31;
  const int wave = tid >> 5;
  const int lg   = lane & 15;                            // N-in-tile / M row sel
  const int hi   = lane >> 4;                            // lane half (K group)
  const int rowbase = blockIdx.x * RPB + wave * 16;
  _Float16* h1s  = lds + wave * (16 * SEC);              // wave-private strip
  _Float16* bst0 = lds + WPB * 16 * SEC;                 // B tile buffer 0
  _Float16* bst1 = bst0 + 16 * SEC;                      // B tile buffer 1

  // Preload GEMM1 A-fragments (16 rows x 256 of x) into 64 VGPRs.
  v16h afrag1[8];
  {
    const _Float16* abase = xf + (size_t)(rowbase + lg) * LATENT + hi * 8;
    #pragma unroll
    for (int kk = 0; kk < 8; ++kk) afrag1[kk] = a_frag(abase + kk * 32);
  }

  // ---- GEMM1: h1 = relu(x @ W1 + b1) -> wave-private LDS (f16) ----
  // B tile = 16 cols x 256 K = 8 KB contiguous in w1T -> 512 v4u, 2 per thread.
  {
    const v4u* g = (const v4u*)w1T;
    v4u* l = (v4u*)bst0;
    l[tid] = g[tid]; l[tid + 256] = g[tid + 256];
  }
  __syncthreads();
  for (int nt = 0; nt < SEC / 16; ++nt) {
    v4u p0 = {}, p1 = {};
    const bool more = (nt + 1) < SEC / 16;               // block-uniform
    if (more) {
      const v4u* g = (const v4u*)(w1T + (size_t)(nt + 1) * 16 * LATENT);
      p0 = g[tid]; p1 = g[tid + 256];
    }
    const _Float16* bcol = ((nt & 1) ? bst1 : bst0) + lg * LATENT + hi * 8;
    v8f acc = {};
    #pragma unroll
    for (int kk = 0; kk < 8; ++kk) {
      v16h bf = a_frag(bcol + kk * 32);          // 2x ds_read_b128
      acc = __builtin_amdgcn_wmma_f32_16x16x32_f16(false, afrag1[kk], false, bf,
                                                   (short)0, acc, false, false);
    }
    const int nbase = nt * 16;
    const float bias = b1[nbase + lg];
    #pragma unroll
    for (int r = 0; r < 8; ++r) {                        // D: M=r+8*hi, N=lg
      float v = acc[r] + bias;
      v = v > 0.f ? v : 0.f;
      h1s[(r + hi * 8) * SEC + nbase + lg] = (_Float16)v;
    }
    if (more) {                                          // fill other buffer
      v4u* l = (v4u*)((nt & 1) ? bst0 : bst1);
      l[tid] = p0; l[tid + 256] = p1;
    }
    __syncthreads();
  }

  // Layout transform bounce: read the wave's h1 strip back ONCE as the 16
  // GEMM2 A-fragments (128 VGPRs). Same-wave DS ops are in-order.
  v16h afrag2[16];
  {
    const _Float16* arow = h1s + lg * SEC + hi * 8;
    #pragma unroll
    for (int kk = 0; kk < 16; ++kk) afrag2[kk] = a_frag(arow + kk * 32);
  }

  // ---- GEMM2 + fused fc3: s += relu(h1 @ W2 + b2) * W3 ----
  // B tile = 16 cols x 512 K = 16 KB contiguous in w2T -> 1024 v4u, 4/thread.
  float sacc[8] = {0.f, 0.f, 0.f, 0.f, 0.f, 0.f, 0.f, 0.f};
  {
    const v4u* g = (const v4u*)w2T;
    v4u* l = (v4u*)bst0;
    l[tid] = g[tid]; l[tid + 256] = g[tid + 256];
    l[tid + 512] = g[tid + 512]; l[tid + 768] = g[tid + 768];
  }
  __syncthreads();
  for (int nt = 0; nt < THIRD / 16; ++nt) {
    v4u p0 = {}, p1 = {}, p2 = {}, p3 = {};
    const bool more = (nt + 1) < THIRD / 16;             // block-uniform
    if (more) {
      const v4u* g = (const v4u*)(w2T + (size_t)(nt + 1) * 16 * SEC);
      p0 = g[tid]; p1 = g[tid + 256]; p2 = g[tid + 512]; p3 = g[tid + 768];
    }
    if (nt + 2 < THIRD / 16)                             // global_prefetch_b8
      __builtin_prefetch(w2T + (size_t)(nt + 2) * 16 * SEC, 0, 1);

    const _Float16* bcol = ((nt & 1) ? bst1 : bst0) + lg * SEC + hi * 8;
    v8f acc = {};
    #pragma unroll
    for (int kk = 0; kk < 16; ++kk) {
      v16h bf = a_frag(bcol + kk * 32);          // 2x ds_read_b128
      acc = __builtin_amdgcn_wmma_f32_16x16x32_f16(false, afrag2[kk], false, bf,
                                                   (short)0, acc, false, false);
    }
    const int nbase = nt * 16;
    const float bias = b2[nbase + lg];
    const float w3v  = w3[nbase + lg];
    #pragma unroll
    for (int r = 0; r < 8; ++r) {
      float v = acc[r] + bias;
      v = v > 0.f ? v : 0.f;
      sacc[r] += v * w3v;                                // fc3 fused per lane
    }
    if (more) {
      v4u* l = (v4u*)((nt & 1) ? bst0 : bst1);
      l[tid] = p0; l[tid + 256] = p1; l[tid + 512] = p2; l[tid + 768] = p3;
    }
    __syncthreads();
  }

  // Reduce N across the 16 lanes of each half-wave (xor stays within halves).
  #pragma unroll
  for (int r = 0; r < 8; ++r) {
    float v = sacc[r];
    v += __shfl_xor(v, 1, 32);
    v += __shfl_xor(v, 2, 32);
    v += __shfl_xor(v, 4, 32);
    v += __shfl_xor(v, 8, 32);
    sacc[r] = v;
  }
  if (lg == 0) {                 // lane 0 -> M=r, lane 16 -> M=8+r
    #pragma unroll
    for (int r = 0; r < 8; ++r) s_out[rowbase + hi * 8 + r] = sacc[r];
  }
}

// ---------------------------------------------------------------------------
// out[t,i,j] = s[t,i] + s[t,j] + b3 : 134 MB stream, NT float4 stores.
// One block per (t,i) row; the 8 KB s row stays hot in cache.
// ---------------------------------------------------------------------------
__launch_bounds__(256)
__global__ void outer_sum_kernel(const float* __restrict__ s,
                                 const float* __restrict__ b3p,
                                 float* __restrict__ out) {
  const int row = blockIdx.x;                 // t*NDIM + i, 0..ROWS-1
  const int t   = row >> 11;
  const float sib = s[row] + b3p[0];
  const v4f* s4 = (const v4f*)(s + ((size_t)t << 11));
  v4f* o4 = (v4f*)(out + (size_t)row * NDIM);
  for (int j4 = threadIdx.x; j4 < NDIM / 4; j4 += 256) {
    v4f v = s4[j4];
    v4f o = { sib + v[0], sib + v[1], sib + v[2], sib + v[3] };
    __builtin_nontemporal_store(o, &o4[j4]);
  }
}

// ---------------------------------------------------------------------------
extern "C" void kernel_launch(void* const* d_in, const int* in_sizes, int n_in,
                              void* d_out, int out_size, void* d_ws, size_t ws_size,
                              hipStream_t stream) {
  const float* x  = (const float*)d_in[0];
  const float* W1 = (const float*)d_in[1];
  const float* b1 = (const float*)d_in[2];
  const float* W2 = (const float*)d_in[3];
  const float* b2 = (const float*)d_in[4];
  const float* W3 = (const float*)d_in[5];
  const float* b3 = (const float*)d_in[6];
  float* out = (float*)d_out;

  // Workspace layout (512B-aligned): xf16 | W1^T f16 | W2^T f16 | s
  char* ws = (char*)d_ws;
  const size_t XF_B  = (size_t)ROWS * LATENT * 2;        // 8 MB
  const size_t W1T_B = (size_t)SEC * LATENT * 2;         // 256 KB
  const size_t W2T_B = (size_t)THIRD * SEC * 2;          // 1 MB
  _Float16* xf  = (_Float16*)(ws);
  _Float16* w1T = (_Float16*)(ws + XF_B);
  _Float16* w2T = (_Float16*)(ws + XF_B + W1T_B);
  float*    sbf = (float*)   (ws + XF_B + W1T_B + W2T_B);

  cvt_x_f16<<<(ROWS * LATENT / 4) / 256, 256, 0, stream>>>(x, xf);
  cvt_w_transpose<<<(SEC * LATENT + 255) / 256, 256, 0, stream>>>(W1, w1T, LATENT, SEC);
  cvt_w_transpose<<<(THIRD * SEC + 255) / 256, 256, 0, stream>>>(W2, w2T, SEC, THIRD);

  // LDS: 8 waves * 16*512 halfs (h1) + 2 * 8192 halfs (B double buffer) = 160 KB
  const size_t lds_bytes = (size_t)(WPB * 16 * SEC + 2 * 16 * SEC) * sizeof(_Float16);
  mlp_s_kernel<<<ROWS / RPB, WPB * 32, lds_bytes, stream>>>(
      xf, w1T, w2T, b1, b2, W3, sbf);

  outer_sum_kernel<<<ROWS, 256, 0, stream>>>(sbf, b3, out);
}
